// GraphTransformer_31817117729187
// MI455X (gfx1250) — matrix operand order repaired and checked by reading, not compiled, not run on hardware
//
#include <hip/hip_runtime.h>
#include <hip/hip_bf16.h>
#include <math.h>

// ---------------- problem constants (match reference) ----------------
#define GN 65536       // nodes
#define GF 128         // in features
#define GD 256         // hidden dim
#define GE (GN * 8)    // edges (without self loops)
#define GCS 32         // chunk size
#define GT ((GN / GCS) + 1)  // tokens incl class token = 2049
#define GDEPTH 8
#define GH 8
#define GNCLS 10

// ---------------- types ----------------
typedef __bf16 bf16_t;
typedef bf16_t v16bf __attribute__((ext_vector_type(16)));
typedef bf16_t v8bf  __attribute__((ext_vector_type(8)));
typedef float  v8f   __attribute__((ext_vector_type(8)));
typedef unsigned int u32x4 __attribute__((ext_vector_type(4)));
typedef int          i32x4 __attribute__((ext_vector_type(4)));
typedef int          i32x8 __attribute__((ext_vector_type(8)));

__device__ __forceinline__ bf16_t f2bf(float f) {
    unsigned u = __builtin_bit_cast(unsigned, f);
    unsigned r = (u + 0x7FFFu + ((u >> 16) & 1u)) >> 16;
    unsigned short s = (unsigned short)r;
    return __builtin_bit_cast(bf16_t, s);
}

__device__ __forceinline__ unsigned f2ord(float f) {
    unsigned u = __builtin_bit_cast(unsigned, f);
    return (u & 0x80000000u) ? ~u : (u | 0x80000000u);
}
__device__ __forceinline__ float ord2f(unsigned u) {
    unsigned v = (u & 0x80000000u) ? (u & 0x7FFFFFFFu) : ~u;
    return __builtin_bit_cast(float, v);
}

// ---------------- TDM availability ----------------
#if defined(__has_builtin)
#if __has_builtin(__builtin_amdgcn_tensor_load_to_lds) && \
    __has_builtin(__builtin_amdgcn_s_wait_tensorcnt)
#define USE_TDM 1
#endif
#endif
#ifndef USE_TDM
#define USE_TDM 0
#endif

// 2D TDM load of bf16 (2-byte) tile (tile1 rows x tile0 cols), row stride
// stride0 elements, into LDS at lds_off with 4 dwords of pad every 16 dwords
// (LDS row stride = 64B data + 16B pad = 80B = 40 bf16).
__device__ __forceinline__ void tdm_load_2d_bf16(
    unsigned lds_off, const void* gaddr,
    unsigned tdim0, unsigned tdim1,
    unsigned tile0, unsigned tile1, unsigned stride0)
{
#if USE_TDM
    unsigned long long ga = (unsigned long long)(size_t)gaddr;
    u32x4 g0;
    g0[0] = 1u;                                        // count=1, user D#
    g0[1] = lds_off;                                   // lds_addr (bytes)
    g0[2] = (unsigned)(ga & 0xFFFFFFFFull);            // global_addr lo
    g0[3] = (unsigned)((ga >> 32) & 0x01FFFFFFull) | (2u << 30); // hi | type=2
    i32x8 g1;
    // data_size=1 (2B), pad_enable=1, pad_interval code 3 (16 dwords),
    // pad_amount code 3 (4 dwords)
    g1[0] = (int)((1u << 16) | (1u << 20) | (3u << 22) | (3u << 25));
    g1[1] = (int)((tdim0 & 0xFFFFu) << 16);                    // abar=0 | tdim0 lo
    g1[2] = (int)((tdim0 >> 16) | ((tdim1 & 0xFFFFu) << 16));  // tdim0 hi | tdim1 lo
    g1[3] = (int)((tdim1 >> 16) | (tile0 << 16));              // tdim1 hi | tile0
    g1[4] = (int)(tile1 & 0xFFFFu);                            // tile1 | tile2=0
    g1[5] = (int)stride0;                                      // dim0 stride lo
    g1[6] = 0;
    g1[7] = 0;
    i32x4 z4 = {0, 0, 0, 0};
#if __clang_major__ >= 23
    i32x8 z8 = {0, 0, 0, 0, 0, 0, 0, 0};
    __builtin_amdgcn_tensor_load_to_lds(g0, g1, z4, z4, z8, 0);
#else
    __builtin_amdgcn_tensor_load_to_lds(g0, g1, z4, z4, 0);
#endif
#else
    (void)lds_off; (void)gaddr; (void)tdim0; (void)tdim1;
    (void)tile0; (void)tile1; (void)stride0;
#endif
}

// =====================================================================
// bf16-WMMA GEMM: C[M,N] = act(A[M,K] @ Bt[N,K]^T + bias) (+res)
// A: bf16 row-major MxK.  Bt: bf16 row-major NxK (pre-transposed weights).
// REQUIRES: K % 32 == 0, N % 64 == 0.
// Block tile 128x64, 8 waves, each wave 32x32 (2x2 WMMA frags).
// Both panels DMA'd to LDS by the Tensor Data Mover (pad -> 80B rows).
// =====================================================================
#define BM 128
#define BN 64
#define BK 32
#define LROW 40   // bf16 elements per padded LDS row (32 data + 8 pad)

__global__ __launch_bounds__(256) void gemm_wmma_bf16(
    const bf16_t* __restrict__ A, const bf16_t* __restrict__ Bt,
    const float* __restrict__ bias, const float* __restrict__ res,
    float* __restrict__ C, int M, int Nn, int K, int act)
{
    __shared__ bf16_t AsB[BM * LROW];   // 10240 B
    __shared__ bf16_t BsB[BN * LROW];   //  5120 B

    const int tid  = threadIdx.x;
    const int lane = tid & 31;
    const int wv   = tid >> 5;            // 8 waves
    const int wm   = wv >> 1;             // 0..3 : 32-row slabs
    const int wn   = wv & 1;              // 0..1 : 32-col slabs
    const int m0   = blockIdx.y * BM;
    const int n0   = blockIdx.x * BN;

    const int lr = lane & 15;
    const int hb = lane >> 4;

    v8f acc00 = {}, acc01 = {}, acc10 = {}, acc11 = {};

    for (int k0 = 0; k0 < K; k0 += BK) {
#if !USE_TDM
        // fallback: cooperative staging of bf16 panels (masked rows for A)
        v8bf a_lo = {}, a_hi = {}, b_lo = {}, b_hi = {};
        {
            int r = tid >> 1;
            int half = (tid & 1) * 16;
            int grow = m0 + r;
            int rc = grow < M ? grow : M - 1;
            const bf16_t* gp = A + (size_t)rc * K + k0 + half;
            v8bf l0 = *(const v8bf*)(gp);
            v8bf l1 = *(const v8bf*)(gp + 8);
            if (grow < M) { a_lo = l0; a_hi = l1; }
        }
        if (tid < 128) {
            int r = tid >> 1;
            int half = (tid & 1) * 16;
            const bf16_t* gp = Bt + (size_t)(n0 + r) * K + k0 + half;
            b_lo = *(const v8bf*)(gp);
            b_hi = *(const v8bf*)(gp + 8);
        }
#endif
        __syncthreads();   // previous iteration's consumers done

#if USE_TDM
        if (wv == 0) {
            tdm_load_2d_bf16((unsigned)(size_t)&AsB[0],
                             A + (size_t)m0 * K + k0,
                             (unsigned)(K - k0), (unsigned)(M - m0),
                             BK, BM, (unsigned)K);
            tdm_load_2d_bf16((unsigned)(size_t)&BsB[0],
                             Bt + (size_t)n0 * K + k0,
                             (unsigned)(K - k0), (unsigned)(Nn - n0),
                             BK, BN, (unsigned)K);
            __builtin_amdgcn_s_wait_tensorcnt(0);
        }
#else
        {
            int r = tid >> 1;
            int half = (tid & 1) * 16;
            *(v8bf*)&AsB[r * LROW + half]     = a_lo;
            *(v8bf*)&AsB[r * LROW + half + 8] = a_hi;
            if (tid < 128) {
                *(v8bf*)&BsB[r * LROW + half]     = b_lo;
                *(v8bf*)&BsB[r * LROW + half + 8] = b_hi;
            }
        }
#endif
        // prefetch next panels while this tile is consumed
        if (k0 + BK < K) {
            int pr = (m0 + (tid >> 1)) < M ? (m0 + (tid >> 1)) : M - 1;
            __builtin_prefetch(A + (size_t)pr * K + k0 + BK, 0, 0);
            __builtin_prefetch(Bt + (size_t)(n0 + (tid & 63)) * K + k0 + BK, 0, 0);
        }
        __syncthreads();

        // ---- fragments: pure ds_load_b128, no conversion ----
        v16bf a0, a1, b0, b1;
        {
            const bf16_t* ap = &AsB[(wm * 32 + lr) * LROW];
            v8bf lo = *(const v8bf*)(ap + hb * 8);
            v8bf hi = *(const v8bf*)(ap + hb * 8 + 16);
            a0 = __builtin_shufflevector(lo, hi, 0,1,2,3,4,5,6,7,8,9,10,11,12,13,14,15);
            ap += 16 * LROW;
            lo = *(const v8bf*)(ap + hb * 8);
            hi = *(const v8bf*)(ap + hb * 8 + 16);
            a1 = __builtin_shufflevector(lo, hi, 0,1,2,3,4,5,6,7,8,9,10,11,12,13,14,15);
        }
        {
            const bf16_t* bp = &BsB[(wn * 32 + lr) * LROW];
            v8bf lo = *(const v8bf*)(bp + hb * 8);
            v8bf hi = *(const v8bf*)(bp + hb * 8 + 16);
            b0 = __builtin_shufflevector(lo, hi, 0,1,2,3,4,5,6,7,8,9,10,11,12,13,14,15);
            bp += 16 * LROW;
            lo = *(const v8bf*)(bp + hb * 8);
            hi = *(const v8bf*)(bp + hb * 8 + 16);
            b1 = __builtin_shufflevector(lo, hi, 0,1,2,3,4,5,6,7,8,9,10,11,12,13,14,15);
        }

        acc00 = __builtin_amdgcn_wmma_f32_16x16x32_bf16(false, a0, false, b0, (short)0, acc00, false, false);
        acc01 = __builtin_amdgcn_wmma_f32_16x16x32_bf16(false, a0, false, b1, (short)0, acc01, false, false);
        acc10 = __builtin_amdgcn_wmma_f32_16x16x32_bf16(false, a1, false, b0, (short)0, acc10, false, false);
        acc11 = __builtin_amdgcn_wmma_f32_16x16x32_bf16(false, a1, false, b1, (short)0, acc11, false, false);

        __syncthreads();
    }

    // ---- epilogue ----
    const int rb = hb * 8;
#pragma unroll
    for (int i = 0; i < 2; i++) {
#pragma unroll
        for (int j = 0; j < 2; j++) {
            v8f acc = (i == 0) ? (j == 0 ? acc00 : acc01)
                               : (j == 0 ? acc10 : acc11);
            int col   = n0 + wn * 32 + j * 16 + lr;
            int rbase = m0 + wm * 32 + i * 16 + rb;
            float bcol = bias ? bias[col] : 0.0f;
#pragma unroll
            for (int e = 0; e < 8; e++) {
                int row = rbase + e;
                if (row < M) {
                    float v = acc[e] + bcol;
                    if (act == 1) v = 0.5f * v * (1.0f + erff(v * 0.70710678118654752f));
                    if (res) v += res[(size_t)row * Nn + col];
                    C[(size_t)row * Nn + col] = v;
                }
            }
        }
    }
}

// ---------------- conversion helpers ----------------
// fp32 -> bf16 row-major copy, 8 elems per thread (n must be multiple of 8)
__global__ __launch_bounds__(256) void cvt_f32_bf16(
    const float* __restrict__ in, bf16_t* __restrict__ out, long long n8)
{
    long long i = (long long)blockIdx.x * 256 + threadIdx.x;
    if (i >= n8) return;
    const float4* p = (const float4*)(in + i * 8);
    float4 f0 = p[0], f1 = p[1];
    v8bf o;
    o[0] = f2bf(f0.x); o[1] = f2bf(f0.y); o[2] = f2bf(f0.z); o[3] = f2bf(f0.w);
    o[4] = f2bf(f1.x); o[5] = f2bf(f1.y); o[6] = f2bf(f1.z); o[7] = f2bf(f1.w);
    *(v8bf*)(out + i * 8) = o;
}

// fp32 RxC -> bf16 CxR transpose (R, C multiples of 32)
__global__ __launch_bounds__(256) void transpose_cvt(
    const float* __restrict__ in, bf16_t* __restrict__ out, int R, int C)
{
    __shared__ float t[32][33];
    int tx = threadIdx.x & 31;
    int ty = threadIdx.x >> 5;     // 0..7
    int c0 = blockIdx.x * 32;
    int r0 = blockIdx.y * 32;
#pragma unroll
    for (int i = 0; i < 32; i += 8)
        t[ty + i][tx] = in[(size_t)(r0 + ty + i) * C + c0 + tx];
    __syncthreads();
#pragma unroll
    for (int i = 0; i < 32; i += 8)
        out[(size_t)(c0 + ty + i) * R + r0 + tx] = f2bf(t[tx][ty + i]);
}

// =====================================================================
// GAT helpers
// =====================================================================
__global__ __launch_bounds__(256) void gat_dots(
    const float* __restrict__ Wh, const float* __restrict__ asrc,
    const float* __restrict__ adst, float* __restrict__ es, float* __restrict__ ed)
{
    __shared__ float r1[256], r2[256];
    int n = blockIdx.x;
    int t = threadIdx.x;
    float w = Wh[(size_t)n * GD + t];
    r1[t] = w * asrc[t];
    r2[t] = w * adst[t];
    __syncthreads();
    for (int s = 128; s > 0; s >>= 1) {
        if (t < s) { r1[t] += r1[t + s]; r2[t] += r2[t + s]; }
        __syncthreads();
    }
    if (t == 0) { es[n] = r1[0]; ed[n] = r2[0]; }
}

__global__ __launch_bounds__(256) void gat_init(
    float* __restrict__ hout, const float* __restrict__ bgat,
    unsigned* __restrict__ emax, float* __restrict__ denom)
{
    long long i = (long long)blockIdx.x * 256 + threadIdx.x;
    long long total = (long long)GN * GD;
    if (i < total) hout[i] = bgat[i & (GD - 1)];
    if (i < GN) { emax[i] = 0u; denom[i] = 0.0f; }
}

__global__ __launch_bounds__(256) void edge_e_max(
    const int* __restrict__ ei, const float* __restrict__ es,
    const float* __restrict__ ed, float* __restrict__ eedge,
    unsigned* __restrict__ emax)
{
    long long i = (long long)blockIdx.x * 256 + threadIdx.x;
    const long long Etot = (long long)GE + GN;
    if (i >= Etot) return;
    int s, d;
    if (i < GE) { s = ei[i]; d = ei[GE + i]; }
    else        { s = (int)(i - GE); d = s; }
    float e = es[s] + ed[d];
    e = (e > 0.0f) ? e : 0.2f * e;   // leaky_relu(0.2)
    eedge[i] = e;
    atomicMax(&emax[d], f2ord(e));
}

__global__ __launch_bounds__(256) void edge_exp_sum(
    const int* __restrict__ ei, float* __restrict__ eedge,
    const unsigned* __restrict__ emax, float* __restrict__ denom)
{
    long long i = (long long)blockIdx.x * 256 + threadIdx.x;
    const long long Etot = (long long)GE + GN;
    if (i >= Etot) return;
    int d = (i < GE) ? ei[GE + i] : (int)(i - GE);
    float ee = expf(eedge[i] - ord2f(emax[d]));
    eedge[i] = ee;
    atomicAdd(&denom[d], ee);
}

#define EPB 8
__global__ __launch_bounds__(256) void edge_scatter(
    const int* __restrict__ ei, const float* __restrict__ eedge,
    const float* __restrict__ denom, const float* __restrict__ Wh,
    float* __restrict__ hout)
{
    const long long Etot = (long long)GE + GN;
    long long base = (long long)blockIdx.x * EPB;
    int t = threadIdx.x;
    for (int j = 0; j < EPB; j++) {
        long long i = base + j;
        if (i >= Etot) return;
        int s, d;
        if (i < GE) { s = ei[i]; d = ei[GE + i]; }
        else        { s = (int)(i - GE); d = s; }
        float alpha = eedge[i] / denom[d];
        atomicAdd(&hout[(size_t)d * GD + t], alpha * Wh[(size_t)s * GD + t]);
    }
}

// =====================================================================
// Transformer helpers
// =====================================================================
__global__ __launch_bounds__(256) void copy_class_tok(
    const float* __restrict__ ct, float* __restrict__ seq)
{
    seq[threadIdx.x] = ct[threadIdx.x];
}

__global__ __launch_bounds__(256) void layernorm_k(
    const float* __restrict__ x, const float* __restrict__ s,
    const float* __restrict__ b, float* __restrict__ y)
{
    __shared__ float red[256];
    int row = blockIdx.x;
    int t = threadIdx.x;
    float v = x[(size_t)row * GD + t];
    red[t] = v;
    __syncthreads();
    for (int st = 128; st > 0; st >>= 1) { if (t < st) red[t] += red[t + st]; __syncthreads(); }
    float mean = red[0] * (1.0f / GD);
    __syncthreads();
    float dv = v - mean;
    red[t] = dv * dv;
    __syncthreads();
    for (int st = 128; st > 0; st >>= 1) { if (t < st) red[t] += red[t + st]; __syncthreads(); }
    float var = red[0] * (1.0f / GD);
    y[(size_t)row * GD + t] = dv * rsqrtf(var + 1e-5f) * s[t] + b[t];
}

// one block per query token; wave w = head w; lane = head-dim element
__global__ __launch_bounds__(256) void attention_k(
    const float* __restrict__ qkv, float* __restrict__ o)
{
    int q = blockIdx.x;
    int h = threadIdx.x >> 5;
    int d = threadIdx.x & 31;
    const float scale = 0.17677669529663687f; // 32^-0.5
    float qd = qkv[(size_t)q * (3 * GD) + h * 32 + d];
    float m = -INFINITY, l = 0.0f, acc = 0.0f;
    for (int t = 0; t < GT; t++) {
        const float* row = qkv + (size_t)t * (3 * GD);
        float kd = row[GD + h * 32 + d];
        float p = qd * kd;
#pragma unroll
        for (int off = 16; off > 0; off >>= 1) p += __shfl_xor(p, off, 32);
        p *= scale;
        float nm = fmaxf(m, p);
        float ef = __expf(m - nm);
        float ep = __expf(p - nm);
        float vd = row[2 * GD + h * 32 + d];
        l   = l * ef + ep;
        acc = acc * ef + ep * vd;
        m = nm;
    }
    o[(size_t)q * GD + h * 32 + d] = acc / l;
}

__global__ __launch_bounds__(256) void classifier_k(
    const float* __restrict__ seq0, const float* __restrict__ Wc,
    const float* __restrict__ bc, float* __restrict__ out)
{
    __shared__ float red[256];
    int t = threadIdx.x;
    for (int c = 0; c < GNCLS; c++) {
        red[t] = seq0[t] * Wc[t * GNCLS + c];
        __syncthreads();
        for (int s = 128; s > 0; s >>= 1) { if (t < s) red[t] += red[t + s]; __syncthreads(); }
        if (t == 0) out[c] = red[0] + bc[c];
        __syncthreads();
    }
}

// =====================================================================
// host orchestration
// =====================================================================
static inline void launch_transpose(const float* in, bf16_t* out, int R, int C,
                                    hipStream_t stream)
{
    dim3 grid(C / 32, R / 32);
    transpose_cvt<<<grid, 256, 0, stream>>>(in, out, R, C);
}

// convert fp32 A (MxK) into abf, then run WMMA GEMM against pre-transposed Bt
static inline void launch_gemm(const float* A_f32, bf16_t* abf, const bf16_t* Bt,
                               const float* bias, const float* res, float* C,
                               int M, int Nn, int K, int act, hipStream_t stream)
{
    long long n8 = ((long long)M * K) / 8;
    cvt_f32_bf16<<<(unsigned)((n8 + 255) / 256), 256, 0, stream>>>(A_f32, abf, n8);
    dim3 grid(Nn / BN, (M + BM - 1) / BM);
    gemm_wmma_bf16<<<grid, 256, 0, stream>>>(abf, Bt, bias, res, C, M, Nn, K, act);
}

extern "C" void kernel_launch(void* const* d_in, const int* in_sizes, int n_in,
                              void* d_out, int out_size, void* d_ws, size_t ws_size,
                              hipStream_t stream)
{
    const float* x        = (const float*)d_in[0];
    const int*   eidx     = (const int*)  d_in[1];
    const float* W_in     = (const float*)d_in[2];
    const float* b_in     = (const float*)d_in[3];
    const float* W_gat    = (const float*)d_in[4];
    const float* a_src    = (const float*)d_in[5];
    const float* a_dst    = (const float*)d_in[6];
    const float* b_gat    = (const float*)d_in[7];
    const float* W_emb    = (const float*)d_in[8];
    const float* b_emb    = (const float*)d_in[9];
    const float* ctok     = (const float*)d_in[10];
    const float* ln1_s    = (const float*)d_in[11];
    const float* ln1_b    = (const float*)d_in[12];
    const float* Wqkv     = (const float*)d_in[13];
    const float* bqkv     = (const float*)d_in[14];
    const float* Wproj    = (const float*)d_in[15];
    const float* bproj    = (const float*)d_in[16];
    const float* ln2_s    = (const float*)d_in[17];
    const float* ln2_b    = (const float*)d_in[18];
    const float* W1       = (const float*)d_in[19];
    const float* b1       = (const float*)d_in[20];
    const float* W2       = (const float*)d_in[21];
    const float* b2       = (const float*)d_in[22];
    const float* W_cls    = (const float*)d_in[23];
    const float* b_cls    = (const float*)d_in[24];
    float* out = (float*)d_out;

    // ---- workspace carve-up (256B aligned blocks) ----
    char* wsb = (char*)d_ws;
    size_t off = 0;
    auto allocb = [&](size_t bytes) {
        void* p = wsb + off;
        off = (off + bytes + 255) & ~(size_t)255;
        return p;
    };
    float*    buf_h   = (float*)allocb((size_t)GN * GD * 4);   // h, then GAT out
    float*    buf_Wh  = (float*)allocb((size_t)GN * GD * 4);
    float*    e_src   = (float*)allocb((size_t)GN * 4);
    float*    e_dst   = (float*)allocb((size_t)GN * 4);
    unsigned* emax    = (unsigned*)allocb((size_t)GN * 4);
    float*    denom   = (float*)allocb((size_t)GN * 4);
    float*    e_edge  = (float*)allocb(((size_t)GE + GN) * 4);
    float*    seq     = (float*)allocb((size_t)GT * GD * 4);
    float*    buf_y   = (float*)allocb((size_t)GT * GD * 4);
    float*    buf_qkv = (float*)allocb((size_t)GT * 3 * GD * 4);
    float*    buf_o   = (float*)allocb((size_t)GT * GD * 4);
    float*    buf_g   = (float*)allocb((size_t)GT * 4 * GD * 4);
    bf16_t*   abf     = (bf16_t*)allocb((size_t)GN * GD * 2);  // bf16 A scratch
    // pre-transposed bf16 weight arena
    bf16_t*   wt_Win  = (bf16_t*)allocb((size_t)GF * GD * 2);          // 256x128
    bf16_t*   wt_Wgat = (bf16_t*)allocb((size_t)GD * GD * 2);          // 256x256
    bf16_t*   wt_Wemb = (bf16_t*)allocb((size_t)GCS * GD * GD * 2);    // 256x8192
    bf16_t*   wt_qkv  = (bf16_t*)allocb((size_t)GDEPTH * GD * 3 * GD * 2);
    bf16_t*   wt_proj = (bf16_t*)allocb((size_t)GDEPTH * GD * GD * 2);
    bf16_t*   wt_w1   = (bf16_t*)allocb((size_t)GDEPTH * GD * 4 * GD * 2);
    bf16_t*   wt_w2   = (bf16_t*)allocb((size_t)GDEPTH * 4 * GD * GD * 2);
    (void)ws_size; (void)n_in; (void)in_sizes; (void)out_size;

    const long long Etot = (long long)GE + GN;

    // 0) transpose + convert all weight matrices (B operands) to bf16 NxK
    launch_transpose(W_in,  wt_Win,  GF, GD, stream);
    launch_transpose(W_gat, wt_Wgat, GD, GD, stream);
    launch_transpose(W_emb, wt_Wemb, GCS * GD, GD, stream);
    for (int i = 0; i < GDEPTH; i++) {
        launch_transpose(Wqkv  + (size_t)i * GD * 3 * GD, wt_qkv  + (size_t)i * GD * 3 * GD, GD, 3 * GD, stream);
        launch_transpose(Wproj + (size_t)i * GD * GD,     wt_proj + (size_t)i * GD * GD,     GD, GD,     stream);
        launch_transpose(W1    + (size_t)i * GD * 4 * GD, wt_w1   + (size_t)i * GD * 4 * GD, GD, 4 * GD, stream);
        launch_transpose(W2    + (size_t)i * 4 * GD * GD, wt_w2   + (size_t)i * 4 * GD * GD, 4 * GD, GD, stream);
    }

    // 1) h = x @ W_in + b_in
    launch_gemm(x, abf, wt_Win, b_in, nullptr, buf_h, GN, GD, GF, 0, stream);
    // 2) Wh = h @ W_gat
    launch_gemm(buf_h, abf, wt_Wgat, nullptr, nullptr, buf_Wh, GN, GD, GD, 0, stream);
    // 3) per-node attention logits
    gat_dots<<<GN, 256, 0, stream>>>(buf_Wh, a_src, a_dst, e_src, e_dst);
    // 4) init GAT accumulators (buf_h reused as output, seeded with b_gat)
    {
        long long total = (long long)GN * GD;
        gat_init<<<(int)((total + 255) / 256), 256, 0, stream>>>(buf_h, b_gat, emax, denom);
    }
    // 5) edge softmax + scatter
    {
        int blocks = (int)((Etot + 255) / 256);
        edge_e_max<<<blocks, 256, 0, stream>>>(eidx, e_src, e_dst, e_edge, emax);
        edge_exp_sum<<<blocks, 256, 0, stream>>>(eidx, e_edge, emax, denom);
        int sblocks = (int)((Etot + EPB - 1) / EPB);
        edge_scatter<<<sblocks, 256, 0, stream>>>(eidx, e_edge, denom, buf_Wh, buf_h);
    }
    // 6) tok = h.reshape(2048, 8192) @ W_emb + b_emb  -> seq rows 1..2048
    launch_gemm(buf_h, abf, wt_Wemb, b_emb, nullptr, seq + GD, GN / GCS, GD, GCS * GD, 0, stream);
    // 7) seq row 0 = class token
    copy_class_tok<<<1, 256, 0, stream>>>(ctok, seq);

    // 8) transformer stack
    for (int i = 0; i < GDEPTH; i++) {
        layernorm_k<<<GT, 256, 0, stream>>>(seq, ln1_s + i * GD, ln1_b + i * GD, buf_y);
        launch_gemm(buf_y, abf, wt_qkv + (size_t)i * GD * 3 * GD,
                    bqkv + (size_t)i * 3 * GD, nullptr, buf_qkv, GT, 3 * GD, GD, 0, stream);
        attention_k<<<GT, 256, 0, stream>>>(buf_qkv, buf_o);
        launch_gemm(buf_o, abf, wt_proj + (size_t)i * GD * GD,
                    bproj + (size_t)i * GD, seq, seq, GT, GD, GD, 0, stream);
        layernorm_k<<<GT, 256, 0, stream>>>(seq, ln2_s + i * GD, ln2_b + i * GD, buf_y);
        launch_gemm(buf_y, abf, wt_w1 + (size_t)i * GD * 4 * GD,
                    b1 + (size_t)i * 4 * GD, nullptr, buf_g, GT, 4 * GD, GD, 1, stream);
        launch_gemm(buf_g, abf, wt_w2 + (size_t)i * 4 * GD * GD,
                    b2 + (size_t)i * GD, seq, seq, GT, GD, 4 * GD, 0, stream);
    }

    // 9) classifier on class token
    classifier_k<<<1, 256, 0, stream>>>(seq, W_cls, b_cls, out);
}
